// LossKMeansWasserstein_34230889349416
// MI455X (gfx1250) — matrix operand order repaired and checked by reading, not compiled, not run on hardware
//
#include <hip/hip_runtime.h>
#include <hip/hip_bf16.h>
#include <math.h>

typedef __attribute__((ext_vector_type(16))) _Float16 v16h;
typedef __attribute__((ext_vector_type(8)))  float    v8f;

#define NPTS  65536
#define DIM   64
#define NCLUS 128
#define EPS_W 1e-8f
#define SORT_CAP 4096
#define LP 68   // LDS row pitch in floats: 272B = 16B-aligned, bank stride 4

// ---------------------------------------------------------------------------
// Kernel 0: zero accumulators (must re-zero every call; graph is replayed)
// ---------------------------------------------------------------------------
__global__ void k_zero(float* __restrict__ fill_acc, float* __restrict__ loss_med) {
    int t = threadIdx.x;
    if (t < NCLUS) fill_acc[t] = 0.f;
    if (t == 0)    loss_med[0] = 0.f;
}

// ---------------------------------------------------------------------------
// Kernel 1: center squared norms cc[k] = ||c_k||^2
// ---------------------------------------------------------------------------
__global__ void k_cc(const float* __restrict__ cen, float* __restrict__ cc) {
    int k = threadIdx.x;            // 1 block x 128 threads
    float s = 0.f;
    #pragma unroll
    for (int d = 0; d < DIM; ++d) { float v = cen[k * DIM + d]; s += v * v; }
    cc[k] = s;
}

// ---------------------------------------------------------------------------
// Kernel 2: dot[i][k] = x_i . c_k via v_wmma_f32_16x16x32_f16 with f16 hi/lo
// split (hi*hi + hi*lo + lo*hi -> ~f32 accuracy). Block = 128 thr (4 waves),
// each wave owns a 16-row stripe x all 128 clusters. Grid = N/64 = 1024.
// Tiles staged with gfx1250 async global->LDS copies (ASYNCcnt path).
// ---------------------------------------------------------------------------
__global__ __launch_bounds__(128) void k_gemm_dot(const float* __restrict__ x,
                                                  const float* __restrict__ cen,
                                                  float* __restrict__ dot) {
    __shared__ float sx[64 * LP];    // 64-row x tile (f32, padded)
    __shared__ float sc[128 * LP];   // all 128 centers (f32, padded)
    const int tid  = threadIdx.x;
    const int row0 = blockIdx.x * 64;

    // Async-stage x tile: 64 rows x 16 chunks of 4 floats (B128 per lane).
    // LDS address operand is the group-segment offset (low 32 bits of flat).
    {
        const float* xg = x + (size_t)row0 * DIM;
        for (int t = tid; t < 64 * 16; t += 128) {
            int r = t >> 4, c4 = (t & 15) << 2;
            unsigned     lds = (unsigned)(uintptr_t)(&sx[r * LP + c4]);
            const float* g   = xg + r * DIM + c4;
            asm volatile("global_load_async_to_lds_b128 %0, %1, off"
                         :: "v"(lds), "v"(g) : "memory");
        }
        for (int t = tid; t < NCLUS * 16; t += 128) {
            int r = t >> 4, c4 = (t & 15) << 2;
            unsigned     lds = (unsigned)(uintptr_t)(&sc[r * LP + c4]);
            const float* g   = cen + r * DIM + c4;
            asm volatile("global_load_async_to_lds_b128 %0, %1, off"
                         :: "v"(lds), "v"(g) : "memory");
        }
        asm volatile("s_wait_asynccnt 0" ::: "memory");
    }
    __syncthreads();

    const int lane = tid & 31;
    const int half = lane >> 4;      // lane group g (0: lanes 0-15, 1: 16-31)
    const int sub  = lane & 15;      // M for A / N for B,C
    const int rowbase = (tid >> 5) * 16;

    // A fragments: 16-bit A 16x32 layout (ISA 7.12.2):
    //   lane holds row M=sub; VGPR v holds K = (v>>2)*16 + g*8 + (v&3)*2 (+1)
    v16h a_hi[2], a_lo[2];
    #pragma unroll
    for (int ks = 0; ks < 2; ++ks) {
        #pragma unroll
        for (int v = 0; v < 8; ++v) {
            int kb = ks * 32 + ((v >> 2) << 4) + (half << 3) + ((v & 3) << 1);
            float f0 = sx[(rowbase + sub) * LP + kb];
            float f1 = sx[(rowbase + sub) * LP + kb + 1];
            _Float16 h0 = (_Float16)f0, h1 = (_Float16)f1;
            a_hi[ks][2 * v]     = h0;
            a_hi[ks][2 * v + 1] = h1;
            a_lo[ks][2 * v]     = (_Float16)(f0 - (float)h0);
            a_lo[ks][2 * v + 1] = (_Float16)(f1 - (float)h1);
        }
    }

    for (int nt = 0; nt < 8; ++nt) {           // 8 cluster tiles of 16
        v8f c = {};
        #pragma unroll
        for (int ks = 0; ks < 2; ++ks) {       // K-dim = 64 -> 2 steps of 32
            // B fragment: 16-bit B 32x16 layout: lane holds col N=sub;
            // VGPR v holds K = (v>>2)*16 + g*8 + (v&3)*2 (+1). B[kk][n] = centers[n][kk].
            v16h b_hi, b_lo;
            #pragma unroll
            for (int v = 0; v < 8; ++v) {
                int kk = ks * 32 + ((v >> 2) << 4) + (half << 3) + ((v & 3) << 1);
                int n  = nt * 16 + sub;
                float f0 = sc[n * LP + kk];
                float f1 = sc[n * LP + kk + 1];
                _Float16 h0 = (_Float16)f0, h1 = (_Float16)f1;
                b_hi[2 * v]     = h0;
                b_hi[2 * v + 1] = h1;
                b_lo[2 * v]     = (_Float16)(f0 - (float)h0);
                b_lo[2 * v + 1] = (_Float16)(f1 - (float)h1);
            }
            c = __builtin_amdgcn_wmma_f32_16x16x32_f16(false, a_hi[ks], false, b_hi, (short)0, c, false, false);
            c = __builtin_amdgcn_wmma_f32_16x16x32_f16(false, a_hi[ks], false, b_lo, (short)0, c, false, false);
            c = __builtin_amdgcn_wmma_f32_16x16x32_f16(false, a_lo[ks], false, b_hi, (short)0, c, false, false);
        }
        #pragma unroll
        for (int j = 0; j < 8; ++j) {          // C layout: M=j+8*g, N=sub
            int rr  = row0 + rowbase + j + 8 * half;
            int col = nt * 16 + sub;
            dot[(size_t)rr * NCLUS + col] = c[j];
        }
    }
}

// ---------------------------------------------------------------------------
// Kernel 3: per-point row ops: dist, argmin (first-min), soft filling accum
// ---------------------------------------------------------------------------
__global__ __launch_bounds__(256) void k_row(const float* __restrict__ x,
                                             const float* __restrict__ dot,
                                             const float* __restrict__ cc,
                                             float* __restrict__ fill_acc,
                                             int* __restrict__ pred) {
    __shared__ float scc[NCLUS];
    __shared__ float sfill[NCLUS];
    int tid = threadIdx.x;
    if (tid < NCLUS) { scc[tid] = cc[tid]; sfill[tid] = 0.f; }
    __syncthreads();

    int i = blockIdx.x * 256 + tid;
    const float* xr = x + (size_t)i * DIM;
    float xx = 0.f;
    #pragma unroll
    for (int d = 0; d < DIM; ++d) { float v = xr[d]; xx += v * v; }

    const float* dr = dot + (size_t)i * NCLUS;
    float wsum = 0.f, dmin = 3.402823e38f;
    int amin = 0;
    for (int k = 0; k < NCLUS; ++k) {
        float d2   = xx + scc[k] - 2.f * dr[k];
        float dist = sqrtf(fmaxf(d2, 0.f));
        if (dist < dmin) { dmin = dist; amin = k; }   // strict < => first min
        wsum += 1.f / (dist + EPS_W);                 // beta=0 -> exponent -1
    }
    pred[i] = amin;
    float inv = 1.f / wsum;
    for (int k = 0; k < NCLUS; ++k) {
        float d2   = xx + scc[k] - 2.f * dr[k];
        float dist = sqrtf(fmaxf(d2, 0.f));
        atomicAdd(&sfill[k], inv / (dist + EPS_W));
    }
    __syncthreads();
    if (tid < NCLUS) atomicAdd(&fill_acc[tid], sfill[tid]);
}

// ---------------------------------------------------------------------------
// Kernel 4: stable (ascending-index) cluster membership lists for x and
// target. Single block, 256 threads: thread c<128 -> cluster c of pred_x,
// thread c+128 -> cluster c of pred_t. Exactly reproduces ix[:m] ordering.
// ---------------------------------------------------------------------------
__global__ __launch_bounds__(256) void k_collect(const int* __restrict__ predX,
                                                 const int* __restrict__ predT,
                                                 int* __restrict__ sortedX,
                                                 int* __restrict__ sortedT,
                                                 int* __restrict__ countX,
                                                 int* __restrict__ countT,
                                                 int* __restrict__ baseX,
                                                 int* __restrict__ baseT) {
    __shared__ int cX[NCLUS], cT[NCLUS], bX[NCLUS], bT[NCLUS];
    int tid = threadIdx.x;
    bool isT = tid >= NCLUS;
    int  c   = tid & (NCLUS - 1);
    const int* pred = isT ? predT : predX;

    int cnt = 0;
    for (int i = 0; i < NPTS; ++i) cnt += (pred[i] == c) ? 1 : 0;
    if (isT) cT[c] = cnt; else cX[c] = cnt;
    __syncthreads();
    if (tid == 0) { int s = 0; for (int k = 0; k < NCLUS; ++k) { bX[k] = s; s += cX[k]; } }
    if (tid == 1) { int s = 0; for (int k = 0; k < NCLUS; ++k) { bT[k] = s; s += cT[k]; } }
    __syncthreads();

    int* dst = isT ? sortedT : sortedX;
    int pos  = isT ? bT[c]   : bX[c];
    for (int i = 0; i < NPTS; ++i)
        if (pred[i] == c) dst[pos++] = i;

    if (!isT) { countX[c] = cX[c]; baseX[c] = bX[c]; }
    else      { countT[c] = cT[c]; baseT[c] = bT[c]; }
}

// ---------------------------------------------------------------------------
// Kernel 5: per-cluster 1D Wasserstein: for each feature, gather the first m
// members' values for x and target into LDS, bitonic-sort (pad to pow2 with
// +INF), sum |diff|; loss += sum / (m*DIM). One block per cluster.
// ---------------------------------------------------------------------------
__global__ __launch_bounds__(256) void k_wass(const float* __restrict__ x,
                                              const float* __restrict__ tgt,
                                              const int* __restrict__ sortedX,
                                              const int* __restrict__ sortedT,
                                              const int* __restrict__ countX,
                                              const int* __restrict__ countT,
                                              const int* __restrict__ baseX,
                                              const int* __restrict__ baseT,
                                              float* __restrict__ loss_acc) {
    __shared__ float sa[SORT_CAP];
    __shared__ float sb[SORT_CAP];
    __shared__ float red[256];
    const int c = blockIdx.x, tid = threadIdx.x;
    int m = min(countX[c], countT[c]);
    m = min(m, SORT_CAP);
    if (m == 0) return;                 // block-uniform
    int P = 1; while (P < m) P <<= 1;
    const int bx = baseX[c], bt = baseT[c];

    float acc = 0.f;
    for (int d = 0; d < DIM; ++d) {
        for (int j = tid; j < P; j += 256) {
            sa[j] = (j < m) ? x[(size_t)sortedX[bx + j] * DIM + d]   : 3.402823e38f;
            sb[j] = (j < m) ? tgt[(size_t)sortedT[bt + j] * DIM + d] : 3.402823e38f;
        }
        __syncthreads();
        for (int k = 2; k <= P; k <<= 1) {
            for (int j = k >> 1; j > 0; j >>= 1) {
                for (int idx = tid; idx < P; idx += 256) {
                    int l = idx ^ j;
                    if (l > idx) {
                        bool asc = ((idx & k) == 0);
                        float va = sa[idx], vb = sa[l];
                        if ((va > vb) == asc) { sa[idx] = vb; sa[l] = va; }
                        float wa = sb[idx], wb = sb[l];
                        if ((wa > wb) == asc) { sb[idx] = wb; sb[l] = wa; }
                    }
                }
                __syncthreads();
            }
        }
        for (int j = tid; j < m; j += 256) acc += fabsf(sa[j] - sb[j]);
        __syncthreads();
    }
    red[tid] = acc;
    __syncthreads();
    for (int s = 128; s > 0; s >>= 1) { if (tid < s) red[tid] += red[tid + s]; __syncthreads(); }
    if (tid == 0) atomicAdd(loss_acc, red[0] / (float)(m * DIM));
}

// ---------------------------------------------------------------------------
// Kernel 6: final scalar: loss_fil (MSE over K of filling diff) + loss_med
// ---------------------------------------------------------------------------
__global__ void k_final(const float* __restrict__ fill_acc,
                        const float* __restrict__ fill_tgt,
                        const float* __restrict__ loss_med,
                        float* __restrict__ out) {
    __shared__ float red[NCLUS];
    int t = threadIdx.x;
    float diff = fill_acc[t] * (1.f / (float)NPTS) - fill_tgt[t];
    red[t] = diff * diff;
    __syncthreads();
    for (int s = 64; s > 0; s >>= 1) { if (t < s) red[t] += red[t + s]; __syncthreads(); }
    if (t == 0) out[0] = red[0] * (1.f / (float)NCLUS) + loss_med[0];
}

// ---------------------------------------------------------------------------
extern "C" void kernel_launch(void* const* d_in, const int* in_sizes, int n_in,
                              void* d_out, int out_size, void* d_ws, size_t ws_size,
                              hipStream_t stream) {
    (void)in_sizes; (void)n_in; (void)out_size; (void)ws_size;
    const float* x      = (const float*)d_in[0];   // [N, 64]
    const float* tgt    = (const float*)d_in[1];   // [N, 64]
    const float* cen    = (const float*)d_in[2];   // [128, 64]
    const int*   predT  = (const int*)  d_in[3];   // [N]
    const float* fillT  = (const float*)d_in[4];   // [128]

    char* ws = (char*)d_ws;
    float* dot      = (float*)(ws);                 // N*128*4 = 33,554,432 B
    int*   predX    = (int*)  (ws + 33554432);      // N*4
    int*   sortedX  = (int*)  (ws + 33816576);      // N*4
    int*   sortedT  = (int*)  (ws + 34078720);      // N*4
    float* fill_acc = (float*)(ws + 34340864);      // 128*4
    float* cc       = (float*)(ws + 34341376);      // 128*4
    int*   countX   = (int*)  (ws + 34341888);
    int*   countT   = (int*)  (ws + 34342400);
    int*   baseX    = (int*)  (ws + 34342912);
    int*   baseT    = (int*)  (ws + 34343424);
    float* loss_med = (float*)(ws + 34343936);

    k_zero   <<<1,    128, 0, stream>>>(fill_acc, loss_med);
    k_cc     <<<1,    128, 0, stream>>>(cen, cc);
    k_gemm_dot<<<NPTS / 64, 128, 0, stream>>>(x, cen, dot);
    k_row    <<<NPTS / 256, 256, 0, stream>>>(x, dot, cc, fill_acc, predX);
    k_collect<<<1,    256, 0, stream>>>(predX, predT, sortedX, sortedT,
                                        countX, countT, baseX, baseT);
    k_wass   <<<NCLUS, 256, 0, stream>>>(x, tgt, sortedX, sortedT,
                                         countX, countT, baseX, baseT, loss_med);
    k_final  <<<1,    128, 0, stream>>>(fill_acc, fillT, loss_med, (float*)d_out);
}
